// FlowPatch_51548197486875
// MI455X (gfx1250) — matrix operand order repaired and checked by reading, not compiled, not run on hardware
//
#include <hip/hip_runtime.h>
#include <hip/hip_bf16.h>
#include <cstdint>

// ---------------------------------------------------------------------------
// FlowPatch (6x affine coupling, per-token MLP 192->256->256->128) on gfx1250.
// bf16 WMMA (f32 accumulate) for all GEMMs; double-buffered async global->LDS
// DMA (ASYNCcnt) overlapped with WMMA; z resident in LDS across layers; the
// feature flip is pure index arithmetic.
// ---------------------------------------------------------------------------

typedef __bf16 bf16_t;
typedef __attribute__((ext_vector_type(8)))  __bf16 v8bf;
typedef __attribute__((ext_vector_type(16))) __bf16 v16bf;
typedef __attribute__((ext_vector_type(8)))  float  v8f;

constexpr int LYR = 6;
constexpr int D   = 128;
constexpr int CND = 128;
constexpr int HID = 256;
constexpr int DIN = D / 2 + CND;   // 192

// LDS row strides (padded, 16B aligned for ds_load_b128)
constexpr int ZS = 132;   // zbuf stride in floats  (528 B)
constexpr int AS = 264;   // abuf stride in bf16    (528 B)
constexpr int WS = 264;   // wslab stride in bf16   (528 B)

constexpr int OFF_Z  = 0;                        // 128*132*4 = 67584
constexpr int OFF_A  = OFF_Z + 128 * ZS * 4;     // + 128*264*2 = 67584
constexpr int OFF_W0 = OFF_A + 128 * AS * 2;     // 135168 (+32*264*2 = 16896)
constexpr int OFF_W1 = OFF_W0 + 32 * WS * 2;     // 152064 (+16896)
constexpr int OFF_C  = OFF_W1 + 32 * WS * 2;     // 168960 (+256)
constexpr int OFF_B  = OFF_C + 128 * 2;          // 169216 (+640*4)
constexpr int OFF_R  = OFF_B + 640 * 4;          // 171776 (+256*4)
constexpr int SMEM_TOTAL = OFF_R + 256 * 4;      // 172800 bytes (< 320KB/WG)

__device__ inline bf16_t f2bf(float f) {
  union { float f; uint32_t u; } v; v.f = f;
  uint32_t r = v.u + 0x7FFFu + ((v.u >> 16) & 1u);   // round-to-nearest-even
  union { unsigned short s; bf16_t b; } o; o.s = (unsigned short)(r >> 16);
  return o.b;
}

__device__ inline float fast_tanh(float v) {
#if __has_builtin(__builtin_amdgcn_tanhf)
  return __builtin_amdgcn_tanhf(v);    // gfx1250 v_tanh_f32
#else
  return tanhf(v);
#endif
}

// ---- CDNA5 async global->LDS DMA (ASYNCcnt-tracked) -----------------------
__device__ inline void async_b128(uint32_t lds_off, const void* g) {
  asm volatile("global_load_async_to_lds_b128 %0, %1, off"
               :: "v"(lds_off), "v"(g) : "memory");
}
__device__ inline void wait_async() {
  asm volatile("s_wait_asynccnt 0" ::: "memory");
}

// ---- WMMA operand builders (per ISA 16-bit A / B VGPR layouts) ------------
// A 16x32 bf16: lane m=l&15; elems 0..7 = K (l>>4)*8..+7, elems 8..15 = K+16.
__device__ inline v16bf load_a_tile(const bf16_t* arow, int kt, int lane) {
  const int khalf = (lane >> 4) << 3;
  v8bf lo = *(const v8bf*)(arow + kt * 32 + khalf);
  v8bf hi = *(const v8bf*)(arow + kt * 32 + 16 + khalf);
  return __builtin_shufflevector(lo, hi, 0,1,2,3,4,5,6,7,8,9,10,11,12,13,14,15);
}
// B 32x16 bf16: lane n=l&15; elems 0..7 = K (l>>4)*8..+7, elems 8..15 = K+16.
__device__ inline v16bf load_b_tile(const bf16_t* wslab, int nt, int lane) {
  const bf16_t* p = wslab + nt * 16 + (lane & 15);
  const int khalf = (lane >> 4) << 3;
  v16bf b;
#pragma unroll
  for (int s = 0; s < 8; ++s) {
    b[s]     = p[(khalf + s) * WS];
    b[8 + s] = p[(16 + khalf + s) * WS];
  }
  return b;
}

// Issue one 32xN weight slab into an LDS buffer (async bf16 DMA, or fp32+cvt).
template<bool PRECONV, int N>
__device__ inline void wslab_issue(bf16_t* wbuf, const bf16_t* wl_bf,
                                   const float* wl_f, int kt, int tid) {
  if constexpr (PRECONV) {
    constexpr int SEGS = N / 8;                       // 16B segments per row
    for (int idx = tid; idx < 32 * SEGS; idx += 256) {
      const int r = idx / SEGS, s = idx % SEGS;
      async_b128((uint32_t)reinterpret_cast<uintptr_t>(wbuf + r * WS + s * 8),
                 (const void*)(wl_bf + (size_t)(kt * 32 + r) * N + s * 8));
    }
  } else {
    constexpr int Q = N / 4;                          // float4 per row
    for (int idx = tid; idx < 32 * Q; idx += 256) {
      const int r = idx / Q, s = idx % Q;
      const float4 v = *(const float4*)(wl_f + (size_t)(kt * 32 + r) * N + s * 4);
      bf16_t* dst = wbuf + r * WS + s * 4;
      dst[0] = f2bf(v.x); dst[1] = f2bf(v.y); dst[2] = f2bf(v.z); dst[3] = f2bf(v.w);
    }
  }
}

// One MLP stage: [16 tokens x 32*KT] @ [32*KT x 16*NT], f32 accumulate.
// Double-buffered weight slabs: DMA for slab kt+1 runs under the WMMAs of kt.
template<bool PRECONV, int KT, int NT>
__device__ inline void mlp_stage(v8f (&c)[NT], const bf16_t* arow,
                                 bf16_t* wbuf0, bf16_t* wbuf1,
                                 const bf16_t* wl_bf, const float* wl_f,
                                 int tid, int lane) {
#pragma unroll
  for (int n = 0; n < NT; ++n)
#pragma unroll
    for (int e = 0; e < 8; ++e) c[n][e] = 0.f;

  // Prologue: previous stage's trailing barrier already freed both buffers.
  wslab_issue<PRECONV, NT * 16>(wbuf0, wl_bf, wl_f, 0, tid);
  if constexpr (PRECONV) wait_async();
  __syncthreads();

  for (int kt = 0; kt < KT; ++kt) {
    bf16_t* cur = (kt & 1) ? wbuf1 : wbuf0;
    bf16_t* nxt = (kt & 1) ? wbuf0 : wbuf1;
    if (kt + 1 < KT)                       // issue DMA for next slab now
      wslab_issue<PRECONV, NT * 16>(nxt, wl_bf, wl_f, kt + 1, tid);

    v16bf a = load_a_tile(arow, kt, lane);
    v16bf bcur = load_b_tile(cur, 0, lane);
#pragma unroll
    for (int n = 0; n < NT; ++n) {         // ping-pong B regs: prefetch n+1
      v16bf bnext = bcur;
      if (n + 1 < NT) bnext = load_b_tile(cur, n + 1, lane);
      c[n] = __builtin_amdgcn_wmma_f32_16x16x32_bf16(false, a, false, bcur,
                                                     (short)0, c[n], false, false);
      bcur = bnext;
    }
    if constexpr (PRECONV) { if (kt + 1 < KT) wait_async(); }
    __syncthreads();                       // slab kt fully consumed; kt+1 ready
  }
}

// Bias + ReLU + bf16 store back into the (wave-private) activation rows.
template<int NT>
__device__ inline void relu_store(const v8f (&c)[NT], bf16_t* abuf,
                                  const float* bias, int rowbase, int lane) {
  const int hi = lane >> 4, nlo = lane & 15;
#pragma unroll
  for (int n = 0; n < NT; ++n) {
    const int col = n * 16 + nlo;
    const float bv = bias[col];
#pragma unroll
    for (int r = 0; r < 8; ++r) {
      float v = c[n][r] + bv;              // C/D layout: M = r + 8*hi, N = nlo
      abuf[(rowbase + r + 8 * hi) * AS + col] = f2bf(v > 0.f ? v : 0.f);
    }
  }
}

__global__ void __launch_bounds__(256)
convert_weights(const float* __restrict__ W1, const float* __restrict__ W2,
                const float* __restrict__ W3, bf16_t* __restrict__ o) {
  const size_t n1 = (size_t)LYR * DIN * HID;
  const size_t n2 = (size_t)LYR * HID * HID;
  const size_t n3 = (size_t)LYR * HID * D;
  size_t i = (size_t)blockIdx.x * 256 + threadIdx.x;
  if (i < n1) o[i] = f2bf(W1[i]);
  else if (i < n1 + n2) o[i] = f2bf(W2[i - n1]);
  else if (i < n1 + n2 + n3) o[i] = f2bf(W3[i - n1 - n2]);
}

template<bool PRECONV>
__global__ void __launch_bounds__(256)
flow_kernel(const float* __restrict__ x, const float* __restrict__ cond,
            const float* __restrict__ W1, const float* __restrict__ b1,
            const float* __restrict__ W2, const float* __restrict__ b2,
            const float* __restrict__ W3, const float* __restrict__ b3,
            const bf16_t* __restrict__ wbf,
            float* __restrict__ outz, float* __restrict__ outld) {
  extern __shared__ char smem[];
  float*  zbuf  = (float*)(smem + OFF_Z);    // [128][ZS] fp32, resident z
  bf16_t* abuf  = (bf16_t*)(smem + OFF_A);   // [128][AS] bf16 activations
  bf16_t* wbuf0 = (bf16_t*)(smem + OFF_W0);  // [32][WS] weight slab (ping)
  bf16_t* wbuf1 = (bf16_t*)(smem + OFF_W1);  // [32][WS] weight slab (pong)
  bf16_t* condb = (bf16_t*)(smem + OFF_C);   // [128] bf16 cond (one batch)
  float*  biasS = (float*)(smem + OFF_B);    // b1(256) b2(256) b3(128)
  float*  red   = (float*)(smem + OFF_R);    // logdet reduction

  const int tid  = threadIdx.x;
  const int lane = tid & 31;
  const int wave = tid >> 5;
  const int rowbase = wave * 16;             // wave-private token rows
  const int bidx = blockIdx.x;               // one batch element per block

  // Stage x[b] (128x128 f32) into zbuf via async DMA; cond -> bf16.
  {
    const float* xb = x + (size_t)bidx * (128 * D);
    for (int idx = tid; idx < 128 * 32; idx += 256) {
      int r = idx >> 5, s = idx & 31;
      async_b128((uint32_t)reinterpret_cast<uintptr_t>(zbuf + r * ZS + s * 4),
                 (const void*)(xb + r * D + s * 4));
    }
    if (tid < CND) condb[tid] = f2bf(cond[(size_t)bidx * CND + tid]);
    wait_async();
  }
  __syncthreads();

  float ldacc = 0.f;
  const size_t w2base = (size_t)LYR * DIN * HID;
  const size_t w3base = w2base + (size_t)LYR * HID * HID;

  for (int l = 0; l < LYR; ++l) {
    __syncthreads();                         // prev-layer biasS readers done
    biasS[tid]       = b1[l * HID + tid];
    biasS[HID + tid] = b2[l * HID + tid];
    if (tid < D) biasS[2 * HID + tid] = b3[l * D + tid];

    // Build stage-1 input [x1 | cond] in bf16 (wave-private rows only).
    for (int j = lane; j < 16 * 96; j += 32) {
      int r = j / 96, kp = (j % 96) * 2;
      bf16_t v0, v1;
      if (kp < 64) {
        v0 = f2bf(zbuf[(rowbase + r) * ZS + kp]);
        v1 = f2bf(zbuf[(rowbase + r) * ZS + kp + 1]);
      } else {
        v0 = condb[kp - 64]; v1 = condb[kp - 63];
      }
      abuf[(rowbase + r) * AS + kp]     = v0;
      abuf[(rowbase + r) * AS + kp + 1] = v1;
    }

    const bf16_t* arow = abuf + (size_t)(rowbase + (lane & 15)) * AS;

    {   // stage 1: [16 x 192] @ [192 x 256] + ReLU
      v8f c[16];
      mlp_stage<PRECONV, 6, 16>(c, arow, wbuf0, wbuf1,
                                wbf + (size_t)l * DIN * HID,
                                W1 + (size_t)l * DIN * HID, tid, lane);
      relu_store<16>(c, abuf, biasS, rowbase, lane);
    }
    {   // stage 2: [16 x 256] @ [256 x 256] + ReLU (in-place; reads precede writes)
      v8f c[16];
      mlp_stage<PRECONV, 8, 16>(c, arow, wbuf0, wbuf1,
                                wbf + w2base + (size_t)l * HID * HID,
                                W2 + (size_t)l * HID * HID, tid, lane);
      relu_store<16>(c, abuf, biasS + HID, rowbase, lane);
    }
    {   // stage 3: [16 x 256] @ [256 x 128] + coupling epilogue + flip
      v8f c[8];
      mlp_stage<PRECONV, 8, 8>(c, arow, wbuf0, wbuf1,
                               wbf + w3base + (size_t)l * HID * D,
                               W3 + (size_t)l * HID * D, tid, lane);
      const int hi = lane >> 4, nlo = lane & 15;
      float x1v[4][8], y2v[4][8];
#pragma unroll
      for (int n = 0; n < 4; ++n) {          // n covers s-cols 0..63
        const int col = n * 16 + nlo;
        const float bs = biasS[2 * HID + col];
        const float bt = biasS[2 * HID + 64 + col];
#pragma unroll
        for (int r = 0; r < 8; ++r) {
          const int tok = rowbase + r + 8 * hi;
          float sv = fast_tanh(c[n][r] + bs) * 0.5f;
          float tv = c[n + 4][r] + bt;       // same lane holds s(n) and t(n)
          float x2 = zbuf[tok * ZS + 64 + col];
          y2v[n][r] = x2 * __expf(sv) + tv;
          x1v[n][r] = zbuf[tok * ZS + col];
          ldacc += sv;
        }
      }
      // Wave-synchronous in-place flip: keep all reads before any writes.
      asm volatile("" ::: "memory");
#pragma unroll
      for (int n = 0; n < 4; ++n) {
        const int col = n * 16 + nlo;
#pragma unroll
        for (int r = 0; r < 8; ++r) {
          const int tok = rowbase + r + 8 * hi;
          zbuf[tok * ZS + (127 - col)] = x1v[n][r];  // z'[64..127] = rev(x1)
          zbuf[tok * ZS + (63  - col)] = y2v[n][r];  // z'[0..63]   = rev(y2)
        }
      }
    }
  }

  // Write z (wave-private rows, float4 vectorized).
  {
    float* oz = outz + (size_t)bidx * (128 * D);
    for (int j = lane; j < 16 * 32; j += 32) {
      int r = j >> 5, s = j & 31;
      *(float4*)(oz + (size_t)(rowbase + r) * D + s * 4) =
          *(const float4*)(zbuf + (size_t)(rowbase + r) * ZS + s * 4);
    }
  }
  // logdet: block reduction over all 8 waves.
  red[tid] = ldacc;
  __syncthreads();
  if (tid < 32) {
    float v = 0.f;
#pragma unroll
    for (int k = 0; k < 8; ++k) v += red[tid + 32 * k];
#pragma unroll
    for (int off = 16; off > 0; off >>= 1) v += __shfl_down(v, off, 32);
    if (tid == 0) outld[bidx] = v;
  }
}

extern "C" void kernel_launch(void* const* d_in, const int* in_sizes, int n_in,
                              void* d_out, int out_size, void* d_ws, size_t ws_size,
                              hipStream_t stream) {
  const float* x  = (const float*)d_in[0];
  const float* cd = (const float*)d_in[1];
  const float* W1 = (const float*)d_in[2];
  const float* b1 = (const float*)d_in[3];
  const float* W2 = (const float*)d_in[4];
  const float* b2 = (const float*)d_in[5];
  const float* W3 = (const float*)d_in[6];
  const float* b3 = (const float*)d_in[7];

  const int B = in_sizes[0] / (128 * D);               // 512
  float* outz  = (float*)d_out;
  float* outld = outz + (size_t)B * 128 * D;

  const size_t WTOT = (size_t)LYR * (DIN * HID + HID * HID + HID * D); // 884736
  const bool preconv = ws_size >= WTOT * sizeof(bf16_t);

  if (preconv) {
    bf16_t* wbf = (bf16_t*)d_ws;
    convert_weights<<<(int)((WTOT + 255) / 256), 256, 0, stream>>>(W1, W2, W3, wbf);
    (void)hipFuncSetAttribute((const void*)flow_kernel<true>,
                              hipFuncAttributeMaxDynamicSharedMemorySize, SMEM_TOTAL);
    flow_kernel<true><<<B, 256, SMEM_TOTAL, stream>>>(x, cd, W1, b1, W2, b2, W3, b3,
                                                      wbf, outz, outld);
  } else {
    (void)hipFuncSetAttribute((const void*)flow_kernel<false>,
                              hipFuncAttributeMaxDynamicSharedMemorySize, SMEM_TOTAL);
    flow_kernel<false><<<B, 256, SMEM_TOTAL, stream>>>(x, cd, W1, b1, W2, b2, W3, b3,
                                                       nullptr, outz, outld);
  }
}
